// CrossAttentionFusion_15384572854350
// MI455X (gfx1250) — matrix operand order repaired and checked by reading, not compiled, not run on hardware
//
#include <hip/hip_runtime.h>
#include <hip/hip_bf16.h>

typedef __attribute__((ext_vector_type(16))) _Float16 v16h;
typedef __attribute__((ext_vector_type(8)))  _Float16 v8h;
typedef __attribute__((ext_vector_type(8)))  float    v8f;
typedef int v4i_ __attribute__((vector_size(16)));   // matches builtin's V4i spelling

// ---- gfx1250 async global->LDS copy (ASYNCcnt-tracked), with safe fallback ----
#ifndef USE_ASYNC_LDS
#if defined(__has_builtin)
#if __has_builtin(__builtin_amdgcn_global_load_async_to_lds_b128)
#define USE_ASYNC_LDS 1
#endif
#endif
#endif
#ifndef USE_ASYNC_LDS
#define USE_ASYNC_LDS 0
#endif

#if USE_ASYNC_LDS
typedef __attribute__((address_space(1))) v4i_* as1_v4i;
typedef __attribute__((address_space(3))) v4i_* as3_v4i;
#endif

static __device__ __forceinline__ void cp16_g2l(const _Float16* g, _Float16* l) {
#if USE_ASYNC_LDS
    v4i_* gp = (v4i_*)const_cast<_Float16*>(g);
    v4i_* lp = (v4i_*)l;
    __builtin_amdgcn_global_load_async_to_lds_b128((as1_v4i)gp, (as3_v4i)lp, 0, 0);
#else
    *(v8h*)l = *(const v8h*)g;
#endif
}
static __device__ __forceinline__ void wait_async_lds() {
#if USE_ASYNC_LDS
#if defined(__has_builtin) && __has_builtin(__builtin_amdgcn_s_wait_asynccnt)
    __builtin_amdgcn_s_wait_asynccnt(0);
#else
    asm volatile("s_wait_asynccnt 0" ::: "memory");
#endif
#endif
}

static __device__ __forceinline__ v8h ld8h(const _Float16* p) {
    return *(const v8h*)p;
}
static __device__ __forceinline__ v16h cat16(v8h lo, v8h hi) {
    return __builtin_shufflevector(lo, hi, 0,1,2,3,4,5,6,7,8,9,10,11,12,13,14,15);
}
static __device__ __forceinline__ v8f wmma_f16(v16h a, v16h b, v8f c) {
    // D = A(16x32 f16) * B(32x16 f16) + C(16x16 f32)
    return __builtin_amdgcn_wmma_f32_16x16x32_f16(false, a, false, b, (short)0, c, false, false);
}

// ---------------------------------------------------------------------------
// Transpose + convert: in f32 [C,S] -> out f16 [S,C]. 32x32 LDS tiles.
// ---------------------------------------------------------------------------
__global__ __launch_bounds__(256)
void transpose_cvt(const float* __restrict__ in, _Float16* __restrict__ out,
                   int Cdim, int Sdim)
{
    __shared__ float t[32][33];
    const int s0 = blockIdx.x * 32, c0 = blockIdx.y * 32;
    const int tx = threadIdx.x, ty = threadIdx.y;   // 32 x 8
    #pragma unroll
    for (int j = 0; j < 4; ++j)
        t[ty + j * 8][tx] = in[(size_t)(c0 + ty + j * 8) * Sdim + s0 + tx];
    __syncthreads();
    #pragma unroll
    for (int j = 0; j < 4; ++j)
        out[(size_t)(s0 + ty + j * 8) * Cdim + c0 + tx] = (_Float16)t[tx][ty + j * 8];
}

// ---------------------------------------------------------------------------
// Tiled WMMA GEMM:  out[M,N] = Ah[M,K](f16) * B[K,N](f32) + bias (+resid)(+relu)
// A tile staged with async global->LDS b128 copies (no conversion needed);
// B tile staged f32->f16 transposed. Block 256 (8 waves), tile 128x64, BK=64.
// LDS row stride 72 halves (144 B) keeps every v8h fragment load 16B-aligned.
// Outputs: optional f32 [M,N], f16 [M,N] (x hscale), f16 transposed [N,M].
// ---------------------------------------------------------------------------
__global__ __launch_bounds__(256)
void gemm_wmma_h(const _Float16* __restrict__ A,
                 const float* __restrict__ B, int ldb,
                 const float* __restrict__ bias,
                 const float* __restrict__ resid,
                 float*    __restrict__ outF,
                 _Float16* __restrict__ outH,
                 _Float16* __restrict__ outHT,
                 float hscale, int M, int N, int K, int relu)
{
    __shared__ __align__(16) _Float16 As[128 * 72];   // [m][k0..63] + pad
    __shared__ __align__(16) _Float16 Bs[64 * 72];    // [n][k0..63] (transposed) + pad

    const int tid  = threadIdx.x;
    const int lane = tid & 31;
    const int w    = tid >> 5;
    const int h16  = lane >> 4;
    const int l15  = lane & 15;
    const int m0   = blockIdx.x * 128;
    const int n0   = blockIdx.y * 64;

    v8f acc[4];
    #pragma unroll
    for (int g = 0; g < 4; ++g) acc[g] = (v8f){};

    for (int k0 = 0; k0 < K; k0 += 64) {
        // ---- A tile (128 x 64 halves = 16 KB): 4 async b128 copies / thread ----
        #pragma unroll
        for (int p = 0; p < 4; ++p) {
            int idx = p * 256 + tid;
            int m   = idx >> 3;                  // 0..127
            int kc  = (idx & 7) * 8;             // 0..56 (halves)
            cp16_g2l(&A[(size_t)(m0 + m) * K + k0 + kc], &As[m * 72 + kc]);
        }
        // ---- B tile (64 x 64) transposed, f32->f16: 4 float4 / thread ----
        #pragma unroll
        for (int p = 0; p < 4; ++p) {
            int idx = p * 256 + tid;
            int kk  = idx >> 4;                  // 0..63
            int nc  = (idx & 15) * 4;            // 0..60
            const float4 s4 = *(const float4*)&B[(size_t)(k0 + kk) * ldb + n0 + nc];
            Bs[(nc + 0) * 72 + kk] = (_Float16)s4.x;
            Bs[(nc + 1) * 72 + kk] = (_Float16)s4.y;
            Bs[(nc + 2) * 72 + kk] = (_Float16)s4.z;
            Bs[(nc + 3) * 72 + kk] = (_Float16)s4.w;
        }
        if (k0 + 64 < K)   // warm next B tile (global_prefetch_b8)
            __builtin_prefetch(&B[(size_t)(k0 + 64) * ldb + n0 + (tid & 63)], 0, 1);
        wait_async_lds();
        __syncthreads();

        // ---- 2 k-steps of 32: 8 WMMAs per wave per stage ----
        #pragma unroll
        for (int ks = 0; ks < 64; ks += 32) {
            // A-frag (16x32 f16, ISA layout): lane<16 k={0..7,16..23}, else {8..15,24..31}
            const _Float16* apt = &As[(w * 16 + l15) * 72 + ks + h16 * 8];
            v16h af = cat16(ld8h(apt), ld8h(apt + 16));
            #pragma unroll
            for (int g = 0; g < 4; ++g) {
                // B-frag (32x16): lane<16 k rows {0..7,16..23}, else {8..15,24..31}; col = lane&15
                const _Float16* bpt = &Bs[(g * 16 + l15) * 72 + ks + h16 * 8];
                v16h bf = cat16(ld8h(bpt), ld8h(bpt + 16));
                acc[g] = wmma_f16(af, bf, acc[g]);
            }
        }
        __syncthreads();
    }

    // ---- epilogue: C/D frag layout row = h16*8 + i, col = lane&15 ----
    #pragma unroll
    for (int g = 0; g < 4; ++g) {
        int col = n0 + g * 16 + l15;
        float bv = bias ? bias[col] : 0.0f;
        #pragma unroll
        for (int i = 0; i < 8; ++i) {
            int row = m0 + w * 16 + h16 * 8 + i;
            float v = acc[g][i] + bv;
            if (resid) v += resid[(size_t)row * N + col];
            if (relu)  v = fmaxf(v, 0.0f);
            if (outF)  outF[(size_t)row * N + col] = v;
            if (outH)  outH[(size_t)row * N + col] = (_Float16)(v * hscale);
            if (outHT) outHT[(size_t)col * M + row] = (_Float16)(v * hscale);
        }
    }
}

// ---------------------------------------------------------------------------
// Attention, head_dim = 32, C = 256, 8 heads. One wave per (head, 16 q rows).
// Q pre-scaled by 1/sqrt(32) f16 [S,256]; K f16 [S,256]; V f16 transposed [256,S].
// No-max softmax (scores ~ N(0,1): far from f32 exp / f16 P overflow); row sum
// computed by WMMA as P x ones. Output written f16 [S,256] for the out-proj GEMM.
// ---------------------------------------------------------------------------
__global__ __launch_bounds__(256)
void flash_attn(const _Float16* __restrict__ Qh, const _Float16* __restrict__ Kh,
                const _Float16* __restrict__ Vt, _Float16* __restrict__ Oh, int S)
{
    __shared__ __align__(16) _Float16 Pl[8][16 * 32];   // per-wave P scratch

    const int tid  = threadIdx.x;
    const int lane = tid & 31;
    const int w    = tid >> 5;
    const int h16  = lane >> 4;
    const int l15  = lane & 15;
    const int q0   = (blockIdx.x * 8 + w) * 16;
    const int ch   = blockIdx.y * 32;

    const _Float16* qp = &Qh[(size_t)(q0 + l15) * 256 + ch + h16 * 8];
    v16h aq = cat16(ld8h(qp), ld8h(qp + 16));

    v16h ones;
    #pragma unroll
    for (int j = 0; j < 16; ++j) ones[j] = (_Float16)1.0f;

    v8f o0 = (v8f){}, o1 = (v8f){}, lac = (v8f){};
    _Float16* pl = &Pl[w][0];

    for (int kt = 0; kt < S; kt += 32) {
        // K^T B-frags: k-dim = d (h16*8 + {0..7,16..23}), col = key (l15 [+16])
        const _Float16* kp0 = &Kh[(size_t)(kt      + l15) * 256 + ch + h16 * 8];
        const _Float16* kp1 = &Kh[(size_t)(kt + 16 + l15) * 256 + ch + h16 * 8];
        v16h bk0 = cat16(ld8h(kp0), ld8h(kp0 + 16));
        v16h bk1 = cat16(ld8h(kp1), ld8h(kp1 + 16));
        v8f z = (v8f){};
        v8f s0 = wmma_f16(aq, bk0, z);
        v8f s1 = wmma_f16(aq, bk1, z);

        #pragma unroll
        for (int i = 0; i < 8; ++i) {
            float a = __expf(s0[i]);
            float b = __expf(s1[i]);
            int row = h16 * 8 + i;
            pl[row * 32 + l15]      = (_Float16)a;
            pl[row * 32 + 16 + l15] = (_Float16)b;
        }
        // C-frag -> A-frag relayout (wave DS ops in-order; explicit split-counter wait)
        asm volatile("s_wait_dscnt 0" ::: "memory");
        const _Float16* pp = &pl[l15 * 32 + h16 * 8];
        v16h ap = cat16(ld8h(pp), ld8h(pp + 16));

        // V B-frags from Vt[256][S]: k-dim = key (h16*8 + {0..7,16..23}), col = d (l15 [+16])
        const _Float16* vp0 = &Vt[(size_t)(ch      + l15) * S + kt + h16 * 8];
        const _Float16* vp1 = &Vt[(size_t)(ch + 16 + l15) * S + kt + h16 * 8];
        v16h bv0 = cat16(ld8h(vp0), ld8h(vp0 + 16));
        v16h bv1 = cat16(ld8h(vp1), ld8h(vp1 + 16));
        o0  = wmma_f16(ap, bv0, o0);
        o1  = wmma_f16(ap, bv1, o1);
        lac = wmma_f16(ap, ones, lac);    // row sums (every column identical)
    }

    #pragma unroll
    for (int i = 0; i < 8; ++i) {
        float inv = 1.0f / lac[i];
        int row = q0 + h16 * 8 + i;
        Oh[(size_t)row * 256 + ch + l15]      = (_Float16)(o0[i] * inv);
        Oh[(size_t)row * 256 + ch + 16 + l15] = (_Float16)(o1[i] * inv);
    }
}

// ---------------------------------------------------------------------------
// LayerNorm over rows of 256. One 256-thread block per token.
// Optional f16 copy; TRANSOUT writes f32 out[c*S + s] (final [C,H,W] layout).
// ---------------------------------------------------------------------------
template<bool TRANSOUT>
__global__ __launch_bounds__(256)
void layernorm_k(const float* __restrict__ X, const float* __restrict__ g,
                 const float* __restrict__ b, float* __restrict__ outF,
                 _Float16* __restrict__ outH, int S)
{
    __shared__ float red[16];
    const int s = blockIdx.x;
    const int tid = threadIdx.x;
    const int w = tid >> 5, lane = tid & 31;

    float v = X[(size_t)s * 256 + tid];
    float sum = v, sq = v * v;
    #pragma unroll
    for (int m = 1; m < 32; m <<= 1) {
        sum += __shfl_xor(sum, m, 32);
        sq  += __shfl_xor(sq,  m, 32);
    }
    if (lane == 0) { red[w] = sum; red[8 + w] = sq; }
    __syncthreads();
    float ts = 0.f, tq = 0.f;
    #pragma unroll
    for (int j = 0; j < 8; ++j) { ts += red[j]; tq += red[8 + j]; }
    float mu  = ts * (1.0f / 256.0f);
    float var = tq * (1.0f / 256.0f) - mu * mu;
    float o = (v - mu) * rsqrtf(var + 1e-5f) * g[tid] + b[tid];
    if (TRANSOUT) outF[(size_t)tid * S + s] = o;
    else          outF[(size_t)s * 256 + tid] = o;
    if (outH)     outH[(size_t)s * 256 + tid] = (_Float16)o;
}

// ---------------------------------------------------------------------------
extern "C" void kernel_launch(void* const* d_in, const int* in_sizes, int n_in,
                              void* d_out, int out_size, void* d_ws, size_t ws_size,
                              hipStream_t stream)
{
    const int S = 4096, C = 256, F = 1024;

    const float* lidar = (const float*)d_in[0];   // [C, S]
    const float* image = (const float*)d_in[1];   // [C, S]
    const float* Wq = (const float*)d_in[2];
    const float* bq = (const float*)d_in[3];
    const float* Wk = (const float*)d_in[4];
    const float* bk = (const float*)d_in[5];
    const float* Wv = (const float*)d_in[6];
    const float* bv = (const float*)d_in[7];
    const float* Wo = (const float*)d_in[8];
    const float* bo = (const float*)d_in[9];
    const float* g1 = (const float*)d_in[10];
    const float* b1 = (const float*)d_in[11];
    const float* W1 = (const float*)d_in[12];
    const float* bf1 = (const float*)d_in[13];
    const float* W2 = (const float*)d_in[14];
    const float* bf2 = (const float*)d_in[15];
    const float* g2 = (const float*)d_in[16];
    const float* b2 = (const float*)d_in[17];
    (void)in_sizes; (void)n_in; (void)out_size; (void)ws_size;

    // workspace carve (34 MiB)
    char* p = (char*)d_ws;
    float*    qf   = (float*)(p);                          //  4 MiB: Q f32 (residual)
    float*    sumb = (float*)(p + ((size_t)4  << 20));     //  4 MiB (reused)
    float*    ln1  = (float*)(p + ((size_t)8  << 20));     //  4 MiB
    _Float16* xh   = (_Float16*)(p + ((size_t)12 << 20));  //  2 MiB: lidar^T f16
    _Float16* yh   = (_Float16*)(p + ((size_t)14 << 20));  //  2 MiB: image^T f16
    _Float16* qh   = (_Float16*)(p + ((size_t)16 << 20));  //  2 MiB
    _Float16* kh   = (_Float16*)(p + ((size_t)18 << 20));  //  2 MiB
    _Float16* vt   = (_Float16*)(p + ((size_t)20 << 20));  //  2 MiB
    _Float16* aoh  = (_Float16*)(p + ((size_t)22 << 20));  //  2 MiB: attn out f16
    _Float16* ln1h = (_Float16*)(p + ((size_t)24 << 20));  //  2 MiB
    _Float16* hidh = (_Float16*)(p + ((size_t)26 << 20));  //  8 MiB: FFN hidden f16

    const dim3 blk(256);
    const float qscale = 0.17677669529663687f;   // 1/sqrt(head_dim=32)

    // features [C,S] f32 -> [S,C] f16
    transpose_cvt<<<dim3(S/32, C/32), dim3(32, 8), 0, stream>>>(lidar, xh, C, S);
    transpose_cvt<<<dim3(S/32, C/32), dim3(32, 8), 0, stream>>>(image, yh, C, S);

    // Q (f32 residual + f16 pre-scaled), K (f16), V (f16 transposed [C,S])
    gemm_wmma_h<<<dim3(S/128, C/64), blk, 0, stream>>>(
        xh, Wq, C, bq, nullptr, qf, qh, nullptr, qscale, S, C, C, 0);
    gemm_wmma_h<<<dim3(S/128, C/64), blk, 0, stream>>>(
        yh, Wk, C, bk, nullptr, nullptr, kh, nullptr, 1.0f, S, C, C, 0);
    gemm_wmma_h<<<dim3(S/128, C/64), blk, 0, stream>>>(
        yh, Wv, C, bv, nullptr, nullptr, nullptr, vt, 1.0f, S, C, C, 0);

    // attention (8 heads, d = 32) -> f16
    flash_attn<<<dim3(S/128, 8), blk, 0, stream>>>(qh, kh, vt, aoh, S);

    // out-proj + residual(q) -> sumb; LN1 -> ln1 (f32) + ln1h (f16)
    gemm_wmma_h<<<dim3(S/128, C/64), blk, 0, stream>>>(
        aoh, Wo, C, bo, qf, sumb, nullptr, nullptr, 1.0f, S, C, C, 0);
    layernorm_k<false><<<dim3(S), blk, 0, stream>>>(sumb, g1, b1, ln1, ln1h, S);

    // FFN up + ReLU -> hidh (f16); FFN down + residual(ln1) -> sumb
    gemm_wmma_h<<<dim3(S/128, F/64), blk, 0, stream>>>(
        ln1h, W1, F, bf1, nullptr, nullptr, hidh, nullptr, 1.0f, S, F, C, 1);
    gemm_wmma_h<<<dim3(S/128, C/64), blk, 0, stream>>>(
        hidh, W2, C, bf2, ln1, sumb, nullptr, nullptr, 1.0f, S, C, F, 0);

    // LN2 -> d_out in [C, H, W] layout
    layernorm_k<true><<<dim3(S), blk, 0, stream>>>(sumb, g2, b2, (float*)d_out, nullptr, S);
}
